// ImputeFormerSparse_71459665871372
// MI455X (gfx1250) — matrix operand order repaired and checked by reading, not compiled, not run on hardware
//
#include <hip/hip_runtime.h>
#include <hip/hip_bf16.h>

typedef __attribute__((ext_vector_type(16))) _Float16 v16h;
typedef __attribute__((ext_vector_type(8)))  _Float16 v8h;
typedef __attribute__((ext_vector_type(8)))  float    v8f;

// ---------------- workspace layout ----------------
// f16 region: per layer LH halves
//   WcT[64][64], Wl1T[64][64], Wl2T[64][64], WvT[64][64], WoT[64][64],
//   Wff1T[256][64], Wff2T[64][256], A_s[32][32]
constexpr int LH      = 54272;
constexpr int OFF_WC  = 0;
constexpr int OFF_WL1 = 4096;
constexpr int OFF_WL2 = 8192;
constexpr int OFF_WV  = 12288;
constexpr int OFF_WO  = 16384;
constexpr int OFF_WF1 = 20480;
constexpr int OFF_WF2 = 36864;
constexpr int OFF_AS  = 53248;
// f32 region: per layer LF floats: bc, bl1, bl2, bv, bo, bff1[256], bff2
constexpr int LF    = 640;
constexpr int FB_C  = 0;
constexpr int FB_L1 = 64;
constexpr int FB_L2 = 128;
constexpr int FB_V  = 192;
constexpr int FB_O  = 256;
constexpr int FB_F1 = 320;
constexpr int FB_F2 = 576;
constexpr size_t WS_F32_BYTE_OFF = (size_t)3 * LH * 2;   // 325632

__device__ __forceinline__ float gelu_exact(float v) {
    return 0.5f * v * (1.0f + erff(v * 0.70710678118654752f));
}

// store one 16x16 C fragment per the f32 C/D layout (VGPR v -> row mbase+v, col n)
template<int MODE>
__device__ __forceinline__ void store_frag(const v8f& c, int mbase, int n, float bval,
                                           float* dstF, int ldf,
                                           _Float16* dstH, int ldh) {
    #pragma unroll
    for (int v = 0; v < 8; ++v) {
        float val = c[v] + bval;
        int row = mbase + v;
        if constexpr (MODE == 0)      dstF[row * ldf + n] = val;
        else if constexpr (MODE == 1) dstH[row * ldh + n] = (_Float16)val;
        else if constexpr (MODE == 2) dstH[row * ldh + n] = (_Float16)gelu_exact(val);
        else if constexpr (MODE == 3) dstH[row * ldh + n] = (_Float16)fmaxf(val, 0.f);
        else                          dstH[n * ldh + row] = (_Float16)val;
    }
}

// ---------------- WMMA GEMM: M=32 rows, wave-per-N-strip ----------------
// A: [32 x kdim] f16, row stride lda (halves). B: [ncols][kdim] f16, row stride ldb.
// ks-outer loop keeps two M-tile accumulators live so each B fragment is
// fetched from global memory exactly once per wave.
// MODE 0: f32 out + bias      -> dstF (stride ldf)
// MODE 1: f16 out (no act)    -> dstH (stride ldh), row-major
// MODE 2: f16 out + bias,gelu -> dstH row-major
// MODE 3: f16 out + bias,relu -> dstH row-major
// MODE 4: f16 out + bias, TRANSPOSED -> dstH[n*ldh + m]
template<int MODE>
__device__ __forceinline__ void gemm32(const _Float16* A, int lda,
                                       const _Float16* Bm, int ldb,
                                       int ncols, int kdim,
                                       float* dstF, int ldf,
                                       _Float16* dstH, int ldh,
                                       const float* bias) {
    const int wave  = threadIdx.x >> 5;
    const int lane  = threadIdx.x & 31;
    const int l15   = lane & 15;
    const int lhalf = lane & 16;                 // 0 or 16
    const int ntc   = ncols >> 4;
    for (int nt = wave; nt < ntc; nt += 4) {
        const int n  = nt * 16 + l15;
        const float bval = bias ? bias[n] : 0.0f;
        v8f c0 = {0.f, 0.f, 0.f, 0.f, 0.f, 0.f, 0.f, 0.f};
        v8f c1 = {0.f, 0.f, 0.f, 0.f, 0.f, 0.f, 0.f, 0.f};
        for (int ks = 0; ks < (kdim >> 5); ++ks) {
            // B fragment (shared by both M-tiles):
            // lane col = n; lanes 0-15 K=ks*32+0..15, lanes 16-31 K=+16..31
            const _Float16* br = Bm + n * ldb + ks * 32 + lhalf;
            v16h bf = *(const v16h*)br;
            // A fragments: lanes 0-15 -> K = ks*32 + 0..15;
            //              lanes 16-31 -> K = ks*32 + 16..31
            const _Float16* ar0 = A + l15 * lda + ks * 32 + lhalf;
            v8h alo = *(const v8h*)(ar0);
            v8h ahi = *(const v8h*)(ar0 + 8);
            v16h af0;
            #pragma unroll
            for (int i = 0; i < 8; ++i) { af0[i] = alo[i]; af0[i + 8] = ahi[i]; }
            c0 = __builtin_amdgcn_wmma_f32_16x16x32_f16(
                     false, af0, false, bf, (short)0, c0, false, false);
            const _Float16* ar1 = ar0 + 16 * lda;
            v8h blo = *(const v8h*)(ar1);
            v8h bhi = *(const v8h*)(ar1 + 8);
            v16h af1;
            #pragma unroll
            for (int i = 0; i < 8; ++i) { af1[i] = blo[i]; af1[i + 8] = bhi[i]; }
            c1 = __builtin_amdgcn_wmma_f32_16x16x32_f16(
                     false, af1, false, bf, (short)0, c1, false, false);
        }
        const int mb = (lhalf ? 8 : 0);
        store_frag<MODE>(c0, mb,      n, bval, dstF, ldf, dstH, ldh);
        store_frag<MODE>(c1, mb + 16, n, bval, dstF, ldf, dstH, ldh);
    }
}

// LayerNorm(xf + add) over D=64; writes f32 back to xf and f16 mirror to xh.
__device__ __forceinline__ void ln_res(float* xf, const float* add,
                                       const float* g, const float* bb,
                                       _Float16* xh) {
    const int tid = threadIdx.x;
    const int row = tid >> 2, sub = tid & 3;
    float*       xr = xf  + row * 64;
    const float* ar = add + row * 64;
    float vals[16];
    float s = 0.f, s2 = 0.f;
    #pragma unroll
    for (int i = 0; i < 16; ++i) {
        int c = sub * 16 + i;
        float v = xr[c] + ar[c];
        vals[i] = v; s += v; s2 += v * v;
    }
    s  += __shfl_xor(s, 1);  s  += __shfl_xor(s, 2);
    s2 += __shfl_xor(s2, 1); s2 += __shfl_xor(s2, 2);
    float mean = s * (1.0f / 64.0f);
    float var  = s2 * (1.0f / 64.0f) - mean * mean;
    float rs   = rsqrtf(var + 1e-5f);
    #pragma unroll
    for (int i = 0; i < 16; ++i) {
        int c = sub * 16 + i;
        float y = (vals[i] - mean) * rs * g[c] + bb[c];
        xr[c] = y;
        xh[row * 72 + c] = (_Float16)y;
    }
}

// ---------------- main kernel ----------------
struct MainArgs {
    const float *qc, *sc, *sv;        // query/sensor coords, values
    const float *ipw, *ipb, *emb;     // input proj w [5][32], b[32], emb[32][32]
    const float *row1, *rob1, *row2, *rob2;  // readout l1.w [64][64], l1.b, l2.w[64], l2.b[1]
    const float *ln[3][8];            // norm1.g/b, norm2.g/b, ln1.g/b, ln2.g/b
    const _Float16 *wsH;
    const float    *wsF;
    float *out;
};

__global__ __launch_bounds__(128) void impute_main(MainArgs a) {
    __shared__ __align__(32) float     xf[32 * 64];
    __shared__ __align__(32) _Float16  xh[32 * 72];
    __shared__ __align__(32) float     buf1[32 * 64];
    __shared__ __align__(32) _Float16  h16[32 * 264];
    __shared__ __align__(32) _Float16  vT[64 * 48];
    __shared__ float red[64];
    __shared__ float scl[3];

    const int b   = blockIdx.x;
    const int tid = threadIdx.x;
    const float* qc = a.qc + (size_t)b * 3;
    const float* sc = a.sc + (size_t)b * 31 * 3;
    const float* sv = a.sv + (size_t)b * 31;

    // per-axis scale = max_k |sc-qc| clamped at 1e-6
    if (tid < 3) {
        float mx = 1e-6f;
        for (int k = 0; k < 31; ++k) mx = fmaxf(mx, fabsf(sc[k * 3 + tid] - qc[tid]));
        scl[tid] = mx;
    }
    __syncthreads();

    // build x[32][64]: [input_proj(feats) | learnable_embedding], token 0 feats = 0
    for (int idx = tid; idx < 32 * 32; idx += 128) {
        int n = idx >> 5, c = idx & 31;
        float vfeat;
        if (n == 0) vfeat = 0.f;
        else {
            int k = n - 1;
            float r0 = (sc[k * 3 + 0] - qc[0]) / scl[0];
            float r1 = (sc[k * 3 + 1] - qc[1]) / scl[1];
            float r2 = (sc[k * 3 + 2] - qc[2]) / scl[2];
            vfeat = a.ipb[c] + sv[k] * a.ipw[0 * 32 + c] + 1.0f * a.ipw[1 * 32 + c]
                  + r0 * a.ipw[2 * 32 + c] + r1 * a.ipw[3 * 32 + c] + r2 * a.ipw[4 * 32 + c];
        }
        xf[n * 64 + c] = vfeat;
        xh[n * 72 + c] = (_Float16)vfeat;
        float e = a.emb[n * 32 + c];
        xf[n * 64 + 32 + c] = e;
        xh[n * 72 + 32 + c] = (_Float16)e;
    }
    __syncthreads();

    for (int L = 0; L < 3; ++L) {
        const _Float16* wl = a.wsH + (size_t)L * LH;
        const float*    fl = a.wsF + (size_t)L * LF;

        // prefetch this layer's weight block into the cache hierarchy
        {
            const char* wb = (const char*)wl;
            for (int o = tid * 128; o < LH * 2; o += 128 * 128)
                __builtin_prefetch(wb + o, 0, 0);
        }

        // ---- projected-attention layer (collapsed to one linear) ----
        gemm32<0>(xh, 72, wl + OFF_WC, 64, 64, 64, buf1, 64, nullptr, 0, fl + FB_C);
        __syncthreads();
        ln_res(xf, buf1, a.ln[L][0], a.ln[L][1], xh);          // norm1
        __syncthreads();
        gemm32<2>(xh, 72, wl + OFF_WL1, 64, 64, 64, nullptr, 0, h16, 264, fl + FB_L1); // gelu
        __syncthreads();
        gemm32<0>(h16, 264, wl + OFF_WL2, 64, 64, 64, buf1, 64, nullptr, 0, fl + FB_L2);
        __syncthreads();
        ln_res(xf, buf1, a.ln[L][2], a.ln[L][3], xh);          // norm2
        __syncthreads();

        // ---- embedded-attention layer ----
        gemm32<4>(xh, 72, wl + OFF_WV, 64, 64, 64, nullptr, 0, vT, 48, fl + FB_V); // v^T
        __syncthreads();
        gemm32<1>(wl + OFF_AS, 32, vT, 48, 64, 32, nullptr, 0, xh, 72, nullptr);   // A_s @ v
        __syncthreads();
        gemm32<0>(xh, 72, wl + OFF_WO, 64, 64, 64, buf1, 64, nullptr, 0, fl + FB_O);
        __syncthreads();
        ln_res(xf, buf1, a.ln[L][4], a.ln[L][5], xh);          // ln1
        __syncthreads();
        gemm32<3>(xh, 72, wl + OFF_WF1, 64, 256, 64, nullptr, 0, h16, 264, fl + FB_F1); // relu
        __syncthreads();
        gemm32<0>(h16, 264, wl + OFF_WF2, 256, 64, 256, buf1, 64, nullptr, 0, fl + FB_F2);
        __syncthreads();
        ln_res(xf, buf1, a.ln[L][6], a.ln[L][7], xh);          // ln2
        __syncthreads();
    }

    // readout on query token (row 0)
    if (tid < 64) {
        float acc = a.rob1[tid];
        for (int k = 0; k < 64; ++k) acc += xf[k] * a.row1[k * 64 + tid];
        acc = fmaxf(acc, 0.f);
        red[tid] = acc * a.row2[tid];
    }
    __syncthreads();
    if (tid == 0) {
        float s = a.rob2[0];
        for (int i = 0; i < 64; ++i) s += red[i];
        a.out[b] = s;
    }
}

// ---------------- prep kernel ----------------
enum { WVO = 0, BVO, WOO, BOO, WVI, BVI, WOI, BOI,
       WL1, BL1, WL2, BL2, WQK, BQK, WVS, BVS, WOS, BOS,
       WF1, BF1, WF2, BF2 };

struct PrepArgs {
    const float* p[3][22];
    const float* emb;
    _Float16* wsH;
    float*    wsF;
};

__device__ __forceinline__ void mm64(const float* A, const float* B, float* C, int tid) {
    for (int idx = tid; idx < 4096; idx += 256) {
        int r = idx >> 6, c = idx & 63;
        float s = 0.f;
        for (int k = 0; k < 64; ++k) s += A[r * 64 + k] * B[k * 64 + c];
        C[r * 64 + c] = s;
    }
}

__device__ __forceinline__ void tr64(const float* src, _Float16* dst, int tid) {
    for (int idx = tid; idx < 4096; idx += 256) {
        int r = idx >> 6, c = idx & 63;
        dst[c * 64 + r] = (_Float16)src[r * 64 + c];
    }
}

__global__ __launch_bounds__(256) void impute_prep(PrepArgs a) {
    __shared__ float T1[64 * 64];
    __shared__ float T2[64 * 64];
    __shared__ float vec1[64], vec2[64];
    __shared__ float q[32 * 64];
    const int L   = blockIdx.x;
    const int tid = threadIdx.x;
    _Float16* wl = a.wsH + (size_t)L * LH;
    float*    fl = a.wsF + (size_t)L * LF;

    // Wc = Wv_out @ Wo_out @ Wv_in @ Wo_in, stored transposed f16
    mm64(a.p[L][WVO], a.p[L][WOO], T1, tid);
    __syncthreads();
    mm64(T1, a.p[L][WVI], T2, tid);
    __syncthreads();
    for (int idx = tid; idx < 4096; idx += 256) {
        int r = idx >> 6, c = idx & 63;
        float s = 0.f;
        for (int k = 0; k < 64; ++k) s += T2[r * 64 + k] * a.p[L][WOI][k * 64 + c];
        wl[OFF_WC + c * 64 + r] = (_Float16)s;
    }
    // bc = ((bv_out Wo_out + bo_out) Wv_in + bv_in) Wo_in + bo_in
    if (tid < 64) {
        float s = a.p[L][BOO][tid];
        for (int k = 0; k < 64; ++k) s += a.p[L][BVO][k] * a.p[L][WOO][k * 64 + tid];
        vec1[tid] = s;
    }
    __syncthreads();
    if (tid < 64) {
        float s = a.p[L][BVI][tid];
        for (int k = 0; k < 64; ++k) s += vec1[k] * a.p[L][WVI][k * 64 + tid];
        vec2[tid] = s;
    }
    __syncthreads();
    if (tid < 64) {
        float s = a.p[L][BOI][tid];
        for (int k = 0; k < 64; ++k) s += vec2[k] * a.p[L][WOI][k * 64 + tid];
        fl[FB_C + tid] = s;
    }

    // transposed f16 weights
    tr64(a.p[L][WL1], wl + OFF_WL1, tid);
    tr64(a.p[L][WL2], wl + OFF_WL2, tid);
    tr64(a.p[L][WVS], wl + OFF_WV,  tid);
    tr64(a.p[L][WOS], wl + OFF_WO,  tid);
    for (int idx = tid; idx < 64 * 256; idx += 256) {     // ff.l1 [64][256] -> [256][64]
        int r = idx >> 8, c = idx & 255;
        wl[OFF_WF1 + c * 64 + r] = (_Float16)a.p[L][WF1][r * 256 + c];
    }
    for (int idx = tid; idx < 64 * 256; idx += 256) {     // ff.l2 [256][64] -> [64][256]
        int r = idx >> 6, c = idx & 63;
        wl[OFF_WF2 + c * 256 + r] = (_Float16)a.p[L][WF2][r * 64 + c];
    }
    // biases
    if (tid < 64) {
        fl[FB_L1 + tid] = a.p[L][BL1][tid];
        fl[FB_L2 + tid] = a.p[L][BL2][tid];
        fl[FB_V  + tid] = a.p[L][BVS][tid];
        fl[FB_O  + tid] = a.p[L][BOS][tid];
        fl[FB_F2 + tid] = a.p[L][BF2][tid];
    }
    if (tid < 256) fl[FB_F1 + tid] = a.p[L][BF1][tid];

    // A_s = softmax((emb Wqk + b)(emb Wqk + b)^T / 8)
    for (int idx = tid; idx < 32 * 64; idx += 256) {
        int n = idx >> 6, d = idx & 63;
        float s = a.p[L][BQK][d];
        for (int k = 0; k < 32; ++k) s += a.emb[n * 32 + k] * a.p[L][WQK][k * 64 + d];
        q[n * 64 + d] = s;
    }
    __syncthreads();
    if (tid < 32) {
        float row[32]; float mx = -1e30f;
        for (int m = 0; m < 32; ++m) {
            float s = 0.f;
            for (int k = 0; k < 64; ++k) s += q[tid * 64 + k] * q[m * 64 + k];
            s *= 0.125f;
            row[m] = s; mx = fmaxf(mx, s);
        }
        float sum = 0.f;
        for (int m = 0; m < 32; ++m) { float e = __expf(row[m] - mx); row[m] = e; sum += e; }
        float inv = 1.f / sum;
        for (int m = 0; m < 32; ++m) wl[OFF_AS + tid * 32 + m] = (_Float16)(row[m] * inv);
    }
}

// ---------------- launcher ----------------
extern "C" void kernel_launch(void* const* d_in, const int* in_sizes, int n_in,
                              void* d_out, int out_size, void* d_ws, size_t ws_size,
                              hipStream_t stream) {
    (void)n_in; (void)out_size; (void)ws_size;
    auto F = [&](int i) { return (const float*)d_in[i]; };

    // flat input order (setup_inputs() dict order, recursive insertion order):
    // 0 query_coords, 1 sensor_coords, 2 sensor_values,
    // 3 input_proj.w, 4 input_proj.b, 5 learnable_embedding,
    // 6 readout.l1.w, 7 readout.l1.b, 8 readout.l2.w, 9 readout.l2.b,
    // per layer L at pb=10+39L:
    //  t.out_attn q.w,q.b,k.w,k.b,v.w,v.b,o.w,o.b    (pb+0..7)
    //  t.in_attn  q.w,q.b,k.w,k.b,v.w,v.b,o.w,o.b    (pb+8..15)
    //  t.projector (pb+16), t.norm1.g/b (17,18), t.norm2.g/b (19,20)
    //  t.mlp l1.w,l1.b,l2.w,l2.b (21..24)
    //  s.qk.w,qk.b (25,26), s.v.w,v.b (27,28), s.o.w,o.b (29,30)
    //  s.ff l1.w,l1.b,l2.w,l2.b (31..34), s.ln1.g/b (35,36), s.ln2.g/b (37,38)
    const int B = in_sizes[0] / 3;

    PrepArgs pa;
    for (int L = 0; L < 3; ++L) {
        int pb = 10 + 39 * L;
        const int map[22] = { pb + 4,  pb + 5,  pb + 6,  pb + 7,
                              pb + 12, pb + 13, pb + 14, pb + 15,
                              pb + 21, pb + 22, pb + 23, pb + 24,
                              pb + 25, pb + 26, pb + 27, pb + 28,
                              pb + 29, pb + 30, pb + 31, pb + 32,
                              pb + 33, pb + 34 };
        for (int j = 0; j < 22; ++j) pa.p[L][j] = F(map[j]);
    }
    pa.emb = F(5);
    pa.wsH = (_Float16*)d_ws;
    pa.wsF = (float*)((char*)d_ws + WS_F32_BYTE_OFF);

    MainArgs ma;
    ma.qc = F(0); ma.sc = F(1); ma.sv = F(2);
    ma.ipw = F(3); ma.ipb = F(4); ma.emb = F(5);
    ma.row1 = F(6); ma.rob1 = F(7); ma.row2 = F(8); ma.rob2 = F(9);
    for (int L = 0; L < 3; ++L) {
        int pb = 10 + 39 * L;
        ma.ln[L][0] = F(pb + 17); ma.ln[L][1] = F(pb + 18);
        ma.ln[L][2] = F(pb + 19); ma.ln[L][3] = F(pb + 20);
        ma.ln[L][4] = F(pb + 35); ma.ln[L][5] = F(pb + 36);
        ma.ln[L][6] = F(pb + 37); ma.ln[L][7] = F(pb + 38);
    }
    ma.wsH = pa.wsH; ma.wsF = pa.wsF;
    ma.out = (float*)d_out;

    impute_prep<<<3, 256, 0, stream>>>(pa);
    impute_main<<<B, 128, 0, stream>>>(ma);
}
